// DAB_feature_36825049596117
// MI455X (gfx1250) — matrix operand order repaired and checked by reading, not compiled, not run on hardware
//
#include <hip/hip_runtime.h>

typedef unsigned int u32;
typedef unsigned short u16;
typedef unsigned long long u64;
typedef __attribute__((ext_vector_type(16))) __bf16 v16bf;
typedef __attribute__((ext_vector_type(8)))  float  v8f;
typedef __attribute__((ext_vector_type(4)))  unsigned int v4u;
typedef __attribute__((ext_vector_type(8)))  int v8i;
typedef __attribute__((ext_vector_type(4)))  int v4i;

#if defined(__AMDGCN__) && __has_builtin(__builtin_amdgcn_tensor_load_to_lds) && \
    __has_builtin(__builtin_amdgcn_s_wait_tensorcnt)
#define USE_TDM 1
#else
#define USE_TDM 0
#endif

__device__ __forceinline__ u16 f2bf(float f) {
    union { float f; u32 u; } x; x.f = f;
    u32 u = x.u;
    return (u16)((u + 0x7FFFu + ((u >> 16) & 1u)) >> 16);  // RNE
}

union Frag { v16bf v; u32 w[8]; };

// ---------------------------------------------------------------------------
// Generic implicit-GEMM conv via WMMA bf16.
//  - Block: 128 threads (4 waves). Tile: M=64 oc, N=64 pixels, K-chunk=32.
//  - grid.x = ceil(OH*OW/64), grid.y = batch, grid.z = OCpad/64
//  - Wp: packed bf16 weights [OCpad][Kpad], K index = ic*KS*KS + kh*KS + kw
//  - A tile (weights) staged to LDS by the Tensor Data Mover (TDM) when
//    available; B tile (im2col gather + f32->bf16) staged by VALU, overlapped.
//  - epilogue: v = acc (+bias[oc]) (*scale[oc] + shift[oc]) (+res) then act
//    act: 0 none, 1 relu, 2 lrelu(0.1)
// ---------------------------------------------------------------------------
template<int KS>
__global__ __launch_bounds__(128) void conv_wmma(
    const float* __restrict__ in, const u16* __restrict__ Wp,
    const float* __restrict__ bias, const float* __restrict__ scale,
    const float* __restrict__ shift, const float* __restrict__ res,
    float* __restrict__ out,
    int IC, int IH, int IW, int OC, int OH, int OW,
    int stride, int pad, int Kpad, int act_mode)
{
    __shared__ __align__(16) u32 sA[64 * 16];  // 64 rows x 32 bf16
    __shared__ __align__(16) u32 sB[64 * 16];  // 64 rows x 32 bf16

    const int tid  = threadIdx.x;
    const int wave = tid >> 5;
    const int lane = tid & 31;
    const int half = lane >> 4;
    const int lr   = lane & 15;

    const int b      = blockIdx.y;
    const int ocBase = blockIdx.z * 64;
    const int p0     = blockIdx.x * 64;
    const int npix   = OH * OW;
    const int K      = IC * KS * KS;

    const float* inB = in + (size_t)b * IC * IH * IW;

    // B-fill mapping: thread -> (pixels bn, bn+32 ; 8 consecutive kk)
    const int bn  = tid & 31;
    const int bk0 = (tid >> 5) * 8;
    int boh[2], bow[2];
    bool pv[2];
#pragma unroll
    for (int g = 0; g < 2; ++g) {
        const int p = p0 + bn + g * 32;
        pv[g]  = p < npix;
        boh[g] = p / OW;
        bow[g] = p % OW;
    }

    v8f acc[4];
#pragma unroll
    for (int i = 0; i < 4; ++i)
        acc[i] = (v8f){0.f, 0.f, 0.f, 0.f, 0.f, 0.f, 0.f, 0.f};

#if USE_TDM
    const u32 ldsA = (u32)(size_t)(void*)sA;
#endif

    const int ktiles = Kpad >> 5;
    for (int kt = 0; kt < ktiles; ++kt) {
        const int k0 = kt << 5;

        // ---- stage A chunk: 64x32 bf16 tile of packed weights -> LDS
#if USE_TDM
        if (wave == 0) {
            const u64 ga = (u64)(const void*)(Wp + ((size_t)ocBase * Kpad + k0));
            v4u g0;
            g0[0] = 1u;                                     // count=1, user mode
            g0[1] = ldsA;                                   // lds_addr
            g0[2] = (u32)ga;                                // global_addr[31:0]
            g0[3] = (u32)((ga >> 32) & 0x01FFFFFFull) | (2u << 30); // addr[56:32] | type=2
            v8i g1;
            g1[0] = (int)0x00010000;     // data_size = 1 (2 bytes)
            g1[1] = (int)(32u << 16);    // tensor_dim0 = 32
            g1[2] = (int)(64u << 16);    // tensor_dim1 = 64
            g1[3] = (int)(32u << 16);    // tile_dim0 = 32
            g1[4] = (int)64;             // tile_dim1 = 64
            g1[5] = (int)Kpad;           // tensor_dim0_stride = Kpad
            g1[6] = 0;
            g1[7] = 0;
            const v4i z4 = {0, 0, 0, 0};
#if __clang_major__ >= 23
            const v8i z8 = {0, 0, 0, 0, 0, 0, 0, 0};
            __builtin_amdgcn_tensor_load_to_lds(g0, g1, z4, z4, z8, 0);
#else
            __builtin_amdgcn_tensor_load_to_lds(g0, g1, z4, z4, 0);
#endif
        }
#else
        {
            const int i  = tid * 8;
            const int m  = i >> 4;
            const int c0 = i & 15;
            const u32* src = (const u32*)Wp + (size_t)(ocBase + m) * (Kpad >> 1) + (k0 >> 1) + c0;
            u32* dst = sA + m * 16 + c0;
#pragma unroll
            for (int j = 0; j < 8; ++j) dst[j] = src[j];
        }
#endif
        // ---- stage B chunk: 64x32 bf16 (im2col gather + cvt), overlaps TDM
#pragma unroll
        for (int g = 0; g < 2; ++g) {
            union { u16 h[8]; u32 w[4]; } tmp;
#pragma unroll
            for (int j = 0; j < 8; ++j) {
                const int k = k0 + bk0 + j;
                float v = 0.f;
                if (pv[g] && k < K) {
                    int ic, kh, kw;
                    if (KS == 1) { ic = k; kh = 0; kw = 0; }
                    else {
                        ic = k / (KS * KS);
                        const int r2 = k - ic * (KS * KS);
                        kh = r2 / KS; kw = r2 - kh * KS;
                    }
                    const int ih = boh[g] * stride + kh - pad;
                    const int iw = bow[g] * stride + kw - pad;
                    if (ih >= 0 && ih < IH && iw >= 0 && iw < IW)
                        v = inB[(size_t)ic * IH * IW + ih * IW + iw];
                }
                tmp.h[j] = f2bf(v);
            }
            u32* dst = sB + (bn + g * 32) * 16 + (bk0 >> 1);
#pragma unroll
            for (int j = 0; j < 4; ++j) dst[j] = tmp.w[j];
        }
#if USE_TDM
        __builtin_amdgcn_s_wait_tensorcnt(0);
#endif
        __syncthreads();

        // ---- fragments (layouts per CDNA5 ISA: 16-bit A 16x32 / B 32x16)
        Frag a;
        {
            const u32* pa = sA + (wave * 16 + lr) * 16;
#pragma unroll
            for (int j = 0; j < 4; ++j) a.w[j] = pa[half * 4 + j];
#pragma unroll
            for (int j = 0; j < 4; ++j) a.w[4 + j] = pa[8 + half * 4 + j];
        }
#pragma unroll
        for (int nt = 0; nt < 4; ++nt) {
            Frag bf;
            const u32* pb = sB + (nt * 16 + lr) * 16;
#pragma unroll
            for (int j = 0; j < 4; ++j) bf.w[j] = pb[half * 4 + j];
#pragma unroll
            for (int j = 0; j < 4; ++j) bf.w[4 + j] = pb[8 + half * 4 + j];
            acc[nt] = __builtin_amdgcn_wmma_f32_16x16x32_bf16(
                false, a.v, false, bf.v, (short)0, acc[nt], false, false);
        }
        __syncthreads();
    }

    // ---- epilogue: C layout 16x16 f32: lane -> (m = r + 8*half, n = lr)
#pragma unroll
    for (int nt = 0; nt < 4; ++nt) {
        const int p = p0 + nt * 16 + lr;
        if (p >= npix) continue;
#pragma unroll
        for (int r = 0; r < 8; ++r) {
            const int oc = ocBase + wave * 16 + r + half * 8;
            if (oc >= OC) continue;
            float v = acc[nt][r];
            if (bias)  v += bias[oc];
            if (scale) v = v * scale[oc] + shift[oc];
            if (res)   v += res[((size_t)b * OC + oc) * npix + p];
            if (act_mode == 1)      v = v > 0.f ? v : 0.f;
            else if (act_mode == 2) v = v >= 0.f ? v : 0.1f * v;
            out[((size_t)b * OC + oc) * npix + p] = v;
        }
    }
}

// ---------------------------------------------------------------------------
// Weight packing: fp32 OIHW -> bf16 [OCpad][Kpad], zero fill
// ---------------------------------------------------------------------------
__global__ void pack_weights(const float* __restrict__ w, u16* __restrict__ wp,
                             int OC, int K, int Kpad, int total)
{
    const int i = blockIdx.x * blockDim.x + threadIdx.x;
    if (i >= total) return;
    const int oc = i / Kpad, k = i - oc * Kpad;
    float v = (oc < OC && k < K) ? w[(size_t)oc * K + k] : 0.f;
    wp[i] = f2bf(v);
}

// BN fold: scale = g*rsqrt(v+eps), shift = b - m*scale
__global__ void bn_prep(const float* g, const float* bt, const float* m,
                        const float* v, float* scale, float* shift, int n)
{
    const int i = blockIdx.x * blockDim.x + threadIdx.x;
    if (i >= n) return;
    const float s = g[i] * rsqrtf(v[i] + 1e-5f);
    scale[i] = s;
    shift[i] = bt[i] - m[i] * s;
}

// Global average pool: one block per (b,ch)
__global__ void gmean(const float* __restrict__ in, float* __restrict__ f, int HW)
{
    __shared__ float s[128];
    const float* p = in + (size_t)blockIdx.x * HW;
    float acc = 0.f;
    for (int i = threadIdx.x; i < HW; i += 128) acc += p[i];
    s[threadIdx.x] = acc;
    __syncthreads();
    for (int o = 64; o > 0; o >>= 1) {
        if (threadIdx.x < o) s[threadIdx.x] += s[threadIdx.x + o];
        __syncthreads();
    }
    if (threadIdx.x == 0) f[blockIdx.x] = s[0] / (float)HW;
}

// MLP + kernel-prediction head: one block per batch, 256 threads
__global__ __launch_bounds__(256) void mlp_head(
    const float* __restrict__ f,
    const float* __restrict__ mw1, const float* __restrict__ mb1,
    const float* __restrict__ mw2, const float* __restrict__ mb2,
    const float* __restrict__ kw1, const float* __restrict__ kw2,
    float* __restrict__ kern)
{
    __shared__ float sf[256], s1[256], s2[256], sh[128];
    const int b = blockIdx.x, t = threadIdx.x;
    sf[t] = f[b * 256 + t];
    __syncthreads();
    float a = mb1[t];
    for (int j = 0; j < 256; ++j) a += mw1[t * 256 + j] * sf[j];
    s1[t] = a >= 0.f ? a : 0.1f * a;
    __syncthreads();
    float e = mb2[t];
    for (int j = 0; j < 256; ++j) e += mw2[t * 256 + j] * s1[j];
    s2[t] = e;
    __syncthreads();
    if (t < 128) {
        float h = 0.f;
        for (int j = 0; j < 256; ++j) h += kw1[t * 256 + j] * s2[j];
        sh[t] = h >= 0.f ? h : 0.1f * h;
    }
    __syncthreads();
    for (int o = t; o < 576; o += 256) {
        float kv = 0.f;
        for (int j = 0; j < 128; ++j) kv += kw2[o * 128 + j] * sh[j];
        kern[b * 576 + o] = kv;
    }
}

// Dynamic depthwise 3x3 (groups=b*c) + lrelu. grid: (HW/256, C, B)
__global__ void dw_lrelu(const float* __restrict__ x, const float* __restrict__ kern,
                         float* __restrict__ out, int H, int W)
{
    const int b = blockIdx.z, c = blockIdx.y;
    const int p = blockIdx.x * 256 + threadIdx.x;
    const int h = p / W, w = p % W;
    const float* kp = kern + (size_t)(b * 64 + c) * 9;
    float k[9];
#pragma unroll
    for (int i = 0; i < 9; ++i) k[i] = kp[i];
    const float* xp = x + ((size_t)(b * 64 + c)) * H * W;
    float s = 0.f;
#pragma unroll
    for (int kh = 0; kh < 3; ++kh) {
        const int ih = h + kh - 1;
        if (ih < 0 || ih >= H) continue;
#pragma unroll
        for (int kw = 0; kw < 3; ++kw) {
            const int iw = w + kw - 1;
            if (iw < 0 || iw >= W) continue;
            s += k[kh * 3 + kw] * xp[ih * W + iw];
        }
    }
    out[((size_t)(b * 64 + c)) * H * W + p] = s >= 0.f ? s : 0.1f * s;
}

// act = lrelu(dynp + x*att), in place over dynp (float4 vectorized)
__global__ void combine(float4* __restrict__ dynp, const float4* __restrict__ x,
                        const float4* __restrict__ att, size_t n4)
{
    const size_t i = (size_t)blockIdx.x * 256 + threadIdx.x;
    if (i >= n4) return;
    const float4 d = dynp[i], xv = x[i], av = att[i];
    float4 r;
    r.x = d.x + xv.x * av.x; r.x = r.x >= 0.f ? r.x : 0.1f * r.x;
    r.y = d.y + xv.y * av.y; r.y = r.y >= 0.f ? r.y : 0.1f * r.y;
    r.z = d.z + xv.z * av.z; r.z = r.z >= 0.f ? r.z : 0.1f * r.z;
    r.w = d.w + xv.w * av.w; r.w = r.w >= 0.f ? r.w : 0.1f * r.w;
    dynp[i] = r;
}

// ---------------------------------------------------------------------------
extern "C" void kernel_launch(void* const* d_in, const int* in_sizes, int n_in,
                              void* d_out, int out_size, void* d_ws, size_t ws_size,
                              hipStream_t stream)
{
    (void)in_sizes; (void)n_in; (void)out_size; (void)ws_size;
    const float* x    = (const float*)d_in[0];
    const float* y    = (const float*)d_in[1];
    const float* ew1  = (const float*)d_in[2];
    const float* eb1  = (const float*)d_in[3];
    const float* bn1g = (const float*)d_in[4];
    const float* bn1b = (const float*)d_in[5];
    const float* bn1m = (const float*)d_in[6];
    const float* bn1v = (const float*)d_in[7];
    const float* ew2  = (const float*)d_in[8];
    const float* eb2  = (const float*)d_in[9];
    const float* bn2g = (const float*)d_in[10];
    const float* bn2b = (const float*)d_in[11];
    const float* bn2m = (const float*)d_in[12];
    const float* bn2v = (const float*)d_in[13];
    const float* mw1  = (const float*)d_in[14];
    const float* mb1  = (const float*)d_in[15];
    const float* mw2  = (const float*)d_in[16];
    const float* mb2  = (const float*)d_in[17];
    const float* kw1  = (const float*)d_in[18];
    const float* kw2  = (const float*)d_in[19];
    const float* cw   = (const float*)d_in[20];
    const float* cb   = (const float*)d_in[21];
    const float* caw1 = (const float*)d_in[22];
    const float* caw2 = (const float*)d_in[23];
    const float* c1w  = (const float*)d_in[24];
    const float* c1b  = (const float*)d_in[25];
    const float* c2w  = (const float*)d_in[26];
    const float* c2b  = (const float*)d_in[27];
    const float* r1w  = (const float*)d_in[28];
    const float* r1b  = (const float*)d_in[29];
    const float* r2w  = (const float*)d_in[30];
    const float* r2b  = (const float*)d_in[31];

    float* out0 = (float*)d_out;                       // (4,64,256,256)
    float* out2 = out0 + (size_t)4 * 64 * 256 * 256;   // (4,64,128,128)
    float* out3 = out2 + (size_t)4 * 64 * 128 * 128;   // (4,64,64,64)

    // ---- workspace layout
    char* W = (char*)d_ws;
    size_t off = 0;
    auto take = [&](size_t bytes) { size_t o = off; off += (bytes + 255) & ~(size_t)255; return o; };
    float* dw    = (float*)(W + take((size_t)4 * 64 * 256 * 256 * 4)); // depthwise out; reused for u/u2
    float* h1    = (float*)(W + take((size_t)4 * 32 * 256 * 256 * 4)); // CA mid; reused for t/t2
    float* att   = (float*)(W + take((size_t)4 * 64 * 256 * 256 * 4));
    float* dynp  = (float*)(W + take((size_t)4 * 64 * 256 * 256 * 4)); // 1x1 out; becomes act in place
    float* enc1o = (float*)(W + take((size_t)4 * 128 * 32 * 32 * 4));
    float* enc2o = (float*)(W + take((size_t)4 * 256 * 30 * 30 * 4));
    float* fbuf  = (float*)(W + take(4 * 256 * 4));
    float* kern  = (float*)(W + take(4 * 576 * 4));
    float* s1    = (float*)(W + take(128 * 4));
    float* sh1   = (float*)(W + take(128 * 4));
    float* s2    = (float*)(W + take(256 * 4));
    float* sh2   = (float*)(W + take(256 * 4));
    u16* ew1p  = (u16*)(W + take((size_t)128 * 576 * 2));
    u16* ew2p  = (u16*)(W + take((size_t)256 * 1152 * 2));
    u16* cwp   = (u16*)(W + take((size_t)64 * 64 * 2));
    u16* caw1p = (u16*)(W + take((size_t)64 * 64 * 2));   // OCpad 64 (OC=32)
    u16* caw2p = (u16*)(W + take((size_t)64 * 32 * 2));
    u16* c1wp  = (u16*)(W + take((size_t)64 * 576 * 2));
    u16* c2wp  = (u16*)(W + take((size_t)64 * 576 * 2));
    u16* r1wp  = (u16*)(W + take((size_t)64 * 576 * 2));
    u16* r2wp  = (u16*)(W + take((size_t)64 * 576 * 2));
    float* tbuf = h1;   // (4,64,128,128) / (4,64,64,64)
    float* ubuf = dw;

    auto pack = [&](const float* w, u16* wp, int OC, int K) {
        const int Kpad = (K + 31) & ~31, OCpad = (OC + 63) & ~63;
        const int total = OCpad * Kpad;
        pack_weights<<<(total + 255) / 256, 256, 0, stream>>>(w, wp, OC, K, Kpad, total);
    };
    auto conv = [&](const float* in, const u16* wp, const float* bias,
                    const float* scale, const float* shift, const float* resi, float* outp,
                    int IC, int IH, int IW, int OC, int stride, int pad, int ks, int act) {
        const int OH = (IH + 2 * pad - ks) / stride + 1;
        const int OW = (IW + 2 * pad - ks) / stride + 1;
        const int K = IC * ks * ks;
        const int Kpad = (K + 31) & ~31;
        const int OCpad = (OC + 63) & ~63;
        dim3 g((OH * OW + 63) / 64, 4, OCpad / 64);
        if (ks == 3)
            conv_wmma<3><<<g, 128, 0, stream>>>(in, wp, bias, scale, shift, resi, outp,
                                                IC, IH, IW, OC, OH, OW, stride, pad, Kpad, act);
        else
            conv_wmma<1><<<g, 128, 0, stream>>>(in, wp, bias, scale, shift, resi, outp,
                                                IC, IH, IW, OC, OH, OW, stride, pad, Kpad, act);
    };

    // ---- prep: BN fold + weight packing (bf16)
    bn_prep<<<1, 128, 0, stream>>>(bn1g, bn1b, bn1m, bn1v, s1, sh1, 128);
    bn_prep<<<1, 256, 0, stream>>>(bn2g, bn2b, bn2m, bn2v, s2, sh2, 256);
    pack(ew1, ew1p, 128, 64 * 9);
    pack(ew2, ew2p, 256, 128 * 9);
    pack(cw, cwp, 64, 64);
    pack(caw1, caw1p, 32, 64);
    pack(caw2, caw2p, 64, 32);
    pack(c1w, c1wp, 64, 576);
    pack(c2w, c2wp, 64, 576);
    pack(r1w, r1wp, 64, 576);
    pack(r2w, r2wp, 64, 576);

    // ---- encoder(y): conv+BN+lrelu x2, global mean, MLP + kernel head
    conv(y, ew1p, eb1, s1, sh1, nullptr, enc1o, 64, 64, 64, 128, 2, 1, 3, 2);
    conv(enc1o, ew2p, eb2, s2, sh2, nullptr, enc2o, 128, 32, 32, 256, 1, 0, 3, 2);
    gmean<<<4 * 256, 128, 0, stream>>>(enc2o, fbuf, 30 * 30);
    mlp_head<<<4, 256, 0, stream>>>(fbuf, mw1, mb1, mw2, mb2, kw1, kw2, kern);

    // ---- dynamic depthwise + 1x1, channel-attention branch, combine
    dw_lrelu<<<dim3(256, 64, 4), 256, 0, stream>>>(x, kern, dw, 256, 256);
    conv(x, caw1p, nullptr, nullptr, nullptr, nullptr, h1, 64, 256, 256, 32, 1, 0, 1, 2);   // lrelu
    conv(h1, caw2p, nullptr, nullptr, nullptr, nullptr, att, 32, 256, 256, 64, 1, 0, 1, 1); // relu
    conv(dw, cwp, cb, nullptr, nullptr, nullptr, dynp, 64, 256, 256, 64, 1, 0, 1, 0);
    {
        const size_t n4 = (size_t)4 * 64 * 256 * 256 / 4;
        combine<<<(unsigned)((n4 + 255) / 256), 256, 0, stream>>>(
            (float4*)dynp, (const float4*)x, (const float4*)att, n4);
    }

    // ---- DAB body
    conv(dynp, c1wp, c1b, nullptr, nullptr, x, out0, 64, 256, 256, 64, 1, 1, 3, 0); // out = conv + x
    conv(out0, c2wp, c2b, nullptr, nullptr, nullptr, tbuf, 64, 256, 256, 64, 2, 1, 3, 0);
    conv(tbuf, r1wp, r1b, nullptr, nullptr, nullptr, ubuf, 64, 128, 128, 64, 1, 1, 3, 1);
    conv(ubuf, r2wp, r2b, nullptr, nullptr, tbuf, out2, 64, 128, 128, 64, 1, 1, 3, 0);
    conv(out2, c2wp, c2b, nullptr, nullptr, nullptr, tbuf, 64, 128, 128, 64, 2, 1, 3, 0);
    conv(tbuf, r1wp, r1b, nullptr, nullptr, nullptr, ubuf, 64, 64, 64, 64, 1, 1, 3, 1);
    conv(ubuf, r2wp, r2b, nullptr, nullptr, tbuf, out3, 64, 64, 64, 64, 1, 1, 3, 0);
}
